// Agent_8778913153394
// MI455X (gfx1250) — matrix operand order repaired and checked by reading, not compiled, not run on hardware
//
#include <hip/hip_runtime.h>

// ---------------------------------------------------------------------------
// Types for CDNA5 WMMA
// ---------------------------------------------------------------------------
typedef __attribute__((ext_vector_type(16))) __bf16 v16bf;
typedef __attribute__((ext_vector_type(8)))  float  v8f;
typedef __attribute__((ext_vector_type(4)))  unsigned int u32x4;
typedef __attribute__((ext_vector_type(4)))  int v4i;

union FragBF16 {
    u32x4  u[2];
    v16bf  v;
};

// Async global->LDS copy path (gfx1250), guarded so either toolchain compiles.
#if defined(__has_builtin)
# if __has_builtin(__builtin_amdgcn_global_load_async_to_lds_b128)
#  define HAVE_ASYNC_LDS 1
# endif
#endif
#ifndef HAVE_ASYNC_LDS
# define HAVE_ASYNC_LDS 0
#endif

#if HAVE_ASYNC_LDS
# if __has_builtin(__builtin_amdgcn_s_wait_asynccnt)
#  define WAIT_ASYNC(n) __builtin_amdgcn_s_wait_asynccnt(n)
# else
#  define WAIT_ASYNC(n) asm volatile("s_wait_asynccnt " #n ::: "memory")
# endif
#else
# define WAIT_ASYNC(n) ((void)0)
#endif

// float -> bf16 (round to nearest even), stored as u16
__device__ __forceinline__ unsigned short f2bf(float f) {
    unsigned u = __float_as_uint(f);
    u += 0x7fffu + ((u >> 16) & 1u);
    return (unsigned short)(u >> 16);
}

// ---------------------------------------------------------------------------
// 1. Gather embeddings / build bf16 GEMM inputs
//    X  [B,1024] bf16 = [rel_emb | ent_emb | prev_h0]
//    SQ [B,1024] bf16 = [ (h1 later) | ent_emb | query ]
// ---------------------------------------------------------------------------
__global__ void __launch_bounds__(256)
gather_kernel(const int* __restrict__ prev_rel, const int* __restrict__ cur_ent,
              const float* __restrict__ rel_table, const float* __restrict__ ent_table,
              const float* __restrict__ prev_h0, const float* __restrict__ query,
              unsigned short* __restrict__ X, unsigned short* __restrict__ SQ)
{
    int b = blockIdx.x, t = threadIdx.x;
    int r = prev_rel[b], e = cur_ent[b];
    float rv = rel_table[(size_t)r * 256 + t];
    float ev = ent_table[(size_t)e * 256 + t];
    size_t xb = (size_t)b * 1024;
    X[xb + t]        = f2bf(rv);
    X[xb + 256 + t]  = f2bf(ev);
    X[xb + 512 + t]  = f2bf(prev_h0[(size_t)b * 512 + t]);
    X[xb + 768 + t]  = f2bf(prev_h0[(size_t)b * 512 + 256 + t]);
    SQ[xb + 512 + t] = f2bf(ev);
    SQ[xb + 768 + t] = f2bf(query[(size_t)b * 256 + t]);
}

// ---------------------------------------------------------------------------
// 2. Transpose fp32 weights [K,N] (optionally stacked [Ka;K-Ka]) into
//    N-major bf16 Wt [N,K] so WMMA B-fragments are contiguous per lane.
//    Block (32,8), grid (K/32, N/32).
// ---------------------------------------------------------------------------
__global__ void __launch_bounds__(256)
transpose_to_bf16(const float* __restrict__ Wa, const float* __restrict__ Wb,
                  int Ka, int K, int N, unsigned short* __restrict__ out)
{
    __shared__ float tile[32][33];
    int tx = threadIdx.x, ty = threadIdx.y;
    int k0 = blockIdx.x * 32, n0 = blockIdx.y * 32;
#pragma unroll
    for (int j = 0; j < 4; ++j) {
        int k = k0 + ty + j * 8;
        int n = n0 + tx;
        float v = (k < Ka) ? Wa[(size_t)k * N + n] : Wb[(size_t)(k - Ka) * N + n];
        tile[ty + j * 8][tx] = v;
    }
    __syncthreads();
#pragma unroll
    for (int j = 0; j < 4; ++j) {
        int n = n0 + ty + j * 8;
        int k = k0 + tx;
        out[(size_t)n * K + k] = f2bf(tile[tx][ty + j * 8]);
    }
}

// ---------------------------------------------------------------------------
// 3. bf16 WMMA GEMM:  C[M,N] (f32 accum) = A[M,K](bf16) * Bt[N,K](bf16)^T
//    + bias[n], optional relu; writes f32 and/or bf16 outputs.
//    Block = 128 threads = 4 waves. Block tile = 128(M) x 64(N);
//    each wave computes 32(M) x 64(N) = 8 WMMAs per 32-wide k-step.
//    B tile (64 cols x 32 K, 4 KB) is double-buffered in LDS, staged with
//    GLOBAL_LOAD_ASYNC_TO_LDS_B128 when available (ASYNCcnt + s_wait_asynccnt).
//    Grid = (N/64, M/128).
// ---------------------------------------------------------------------------
__global__ void __launch_bounds__(128)
wmma_gemm_bf16(const unsigned short* __restrict__ A,
               const unsigned short* __restrict__ Bt,
               const float* __restrict__ bias,
               float* __restrict__ outF,
               unsigned short* __restrict__ outB,
               int M, int N, int K, int relu)
{
    __shared__ unsigned short ldsB[2][64 * 32];   // 2 x 4 KB, [col][k] bf16

    int lane = threadIdx.x & 31;
    int wave = threadIdx.x >> 5;
    int l16  = lane & 15;
    int hi   = lane >> 4;                 // 0: lanes 0-15, 1: lanes 16-31

    int rowbase = blockIdx.y * 128 + wave * 32;   // this wave's 32 M-rows
    int colbase = blockIdx.x * 64;                // block's 64 N-cols

    // A fragments: rows rowbase+l16 and rowbase+16+l16; K-chunks +hi*8, +hi*8+16
    const unsigned short* aptr0 = A + (size_t)(rowbase + l16) * K + hi * 8;
    const unsigned short* aptr1 = A + (size_t)(rowbase + 16 + l16) * K + hi * 8;

    // B tile staging: thread copies two 16B chunks of the 4 KB tile per k-step
    auto copyB = [&](int buf, int kk) {
#pragma unroll
        for (int q = 0; q < 2; ++q) {
            int c   = threadIdx.x + q * 128;      // chunk 0..255
            int col = c >> 2;                     // 0..63
            int off = (c & 3) * 8;                // ushort offset within 64B row
            const unsigned short* g = Bt + (size_t)(colbase + col) * K + kk + off;
            unsigned short*       l = &ldsB[buf][col * 32 + off];
#if HAVE_ASYNC_LDS
            __builtin_amdgcn_global_load_async_to_lds_b128(
                (__attribute__((address_space(1))) v4i*)g,
                (__attribute__((address_space(3))) v4i*)l,
                0, 0);
#else
            *(u32x4*)l = *(const u32x4*)g;
#endif
        }
    };

    v8f acc0[4] = {v8f{}, v8f{}, v8f{}, v8f{}};
    v8f acc1[4] = {v8f{}, v8f{}, v8f{}, v8f{}};

    copyB(0, 0);
    int buf = 0;
    for (int kk = 0; kk < K; kk += 32, buf ^= 1) {
        int has_next = (kk + 32 < K);
        if (has_next) copyB(buf ^ 1, kk + 32);
#if HAVE_ASYNC_LDS
        if (has_next) { WAIT_ASYNC(2); } else { WAIT_ASYNC(0); }
#endif
        __syncthreads();   // current B tile resident for whole block

        FragBF16 a0, a1;
        a0.u[0] = *(const u32x4*)(aptr0 + kk);
        a0.u[1] = *(const u32x4*)(aptr0 + kk + 16);
        a1.u[0] = *(const u32x4*)(aptr1 + kk);
        a1.u[1] = *(const u32x4*)(aptr1 + kk + 16);

#pragma unroll
        for (int j = 0; j < 4; ++j) {
            FragBF16 bf;
            const unsigned short* lp = &ldsB[buf][(j * 16 + l16) * 32 + hi * 8];
            bf.u[0] = *(const u32x4*)lp;
            bf.u[1] = *(const u32x4*)(lp + 16);
            acc0[j] = __builtin_amdgcn_wmma_f32_16x16x32_bf16(
                          false, a0.v, false, bf.v, (short)0, acc0[j], false, false);
            acc1[j] = __builtin_amdgcn_wmma_f32_16x16x32_bf16(
                          false, a1.v, false, bf.v, (short)0, acc1[j], false, false);
        }
        __syncthreads();   // all waves done reading before buffer is overwritten
    }

    // C/D layout: VGPR v, lanes 0-15 -> M=v, lanes 16-31 -> M=v+8; N = l16
#pragma unroll
    for (int i = 0; i < 2; ++i) {
        int mbase = rowbase + i * 16 + hi * 8;
#pragma unroll
        for (int j = 0; j < 4; ++j) {
            v8f c = i ? acc1[j] : acc0[j];
            int n = colbase + j * 16 + l16;
            float bv = bias ? bias[n] : 0.0f;
#pragma unroll
            for (int v = 0; v < 8; ++v) {
                float val = c[v] + bv;
                if (relu) val = fmaxf(val, 0.0f);
                size_t off = (size_t)(mbase + v) * N + n;
                if (outF) outF[off] = val;
                if (outB) outB[off] = f2bf(val);
            }
        }
    }
}

// ---------------------------------------------------------------------------
// 4. LSTM gate activations. z [B,2048] (bias already added in GEMM),
//    gate order i,f,g,o. Writes h,c (f32 outputs) and h as bf16 into
//    target[:,0:512]; optionally converts h_other into target[:,512:1024].
// ---------------------------------------------------------------------------
__global__ void __launch_bounds__(256)
lstm_act_kernel(const float* __restrict__ Z, const float* __restrict__ prev_c,
                const float* __restrict__ h_other,
                float* __restrict__ h_out, float* __restrict__ c_out,
                unsigned short* __restrict__ target)
{
    int b = blockIdx.x;
#pragma unroll
    for (int q = 0; q < 2; ++q) {
        int j = threadIdx.x + q * 256;
        size_t zb = (size_t)b * 2048;
        float zi = Z[zb + j];
        float zf = Z[zb + 512 + j];
        float zg = Z[zb + 1024 + j];
        float zo = Z[zb + 1536 + j];
        float i = 1.0f / (1.0f + expf(-zi));
        float f = 1.0f / (1.0f + expf(-zf));
        float g = tanhf(zg);
        float o = 1.0f / (1.0f + expf(-zo));
        float c = f * prev_c[(size_t)b * 512 + j] + i * g;
        float h = o * tanhf(c);
        h_out[(size_t)b * 512 + j] = h;
        c_out[(size_t)b * 512 + j] = c;
        target[(size_t)b * 1024 + j] = f2bf(h);
        if (h_other)
            target[(size_t)b * 1024 + 512 + j] = f2bf(h_other[(size_t)b * 512 + j]);
    }
}

// ---------------------------------------------------------------------------
// 5. Scores: score[b,a] = dot(rel_table[nr[b,a]], mlp[b,0:256])
//                       + dot(ent_table[ne[b,a]], mlp[b,256:512])
//    One block per b (mlp row staged in LDS), one wave per action.
// ---------------------------------------------------------------------------
__global__ void __launch_bounds__(256)
scores_kernel(const int* __restrict__ next_rel, const int* __restrict__ next_ent,
              const float* __restrict__ rel_table, const float* __restrict__ ent_table,
              const float* __restrict__ mlp,
              float* __restrict__ prelim, float* __restrict__ scorem)
{
    __shared__ float sm[512];
    int b = blockIdx.x;
    sm[threadIdx.x]       = mlp[(size_t)b * 512 + threadIdx.x];
    sm[threadIdx.x + 256] = mlp[(size_t)b * 512 + 256 + threadIdx.x];
    __syncthreads();

    int lane = threadIdx.x & 31;
    int wave = threadIdx.x >> 5;
    for (int a = wave; a < 256; a += 8) {
        int r = next_rel[(size_t)b * 256 + a];
        int e = next_ent[(size_t)b * 256 + a];
        const float* src;
        int base;
        if (lane < 16) { src = rel_table + (size_t)r * 256 + lane * 16;        base = lane * 16; }
        else           { src = ent_table + (size_t)e * 256 + (lane - 16) * 16; base = 256 + (lane - 16) * 16; }
        float acc = 0.0f;
#pragma unroll
        for (int q = 0; q < 16; q += 4) {
            float4 cv = *(const float4*)(src + q);
            acc += cv.x * sm[base + q]     + cv.y * sm[base + q + 1]
                 + cv.z * sm[base + q + 2] + cv.w * sm[base + q + 3];
        }
#pragma unroll
        for (int off = 16; off; off >>= 1) acc += __shfl_xor(acc, off, 32);
        if (lane == 0) {
            prelim[(size_t)b * 256 + a] = acc;
            scorem[(size_t)b * 256 + a] = (r == 0) ? -99999.0f : acc;
        }
    }
}

// ---------------------------------------------------------------------------
// 6. Threefry2x32 (JAX), Gumbel-argmax categorical, log-softmax, loss.
// ---------------------------------------------------------------------------
__device__ __forceinline__ unsigned rotl32(unsigned x, int d) {
    return (x << d) | (x >> (32 - d));
}

__device__ void threefry2x32(unsigned k0, unsigned k1, unsigned& x0, unsigned& x1) {
    unsigned ks0 = k0, ks1 = k1, ks2 = k0 ^ k1 ^ 0x1BD11BDAu;
    const int r0[4] = {13, 15, 26, 6};
    const int r1[4] = {17, 29, 16, 24};
    x0 += ks0; x1 += ks1;
#pragma unroll
    for (int i = 0; i < 4; ++i) { x0 += x1; x1 = rotl32(x1, r0[i]); x1 ^= x0; }
    x0 += ks1; x1 += ks2 + 1u;
#pragma unroll
    for (int i = 0; i < 4; ++i) { x0 += x1; x1 = rotl32(x1, r1[i]); x1 ^= x0; }
    x0 += ks2; x1 += ks0 + 2u;
#pragma unroll
    for (int i = 0; i < 4; ++i) { x0 += x1; x1 = rotl32(x1, r0[i]); x1 ^= x0; }
    x0 += ks0; x1 += ks1 + 3u;
#pragma unroll
    for (int i = 0; i < 4; ++i) { x0 += x1; x1 = rotl32(x1, r1[i]); x1 ^= x0; }
    x0 += ks1; x1 += ks2 + 4u;
#pragma unroll
    for (int i = 0; i < 4; ++i) { x0 += x1; x1 = rotl32(x1, r0[i]); x1 ^= x0; }
    x0 += ks2; x1 += ks0 + 5u;
}

__global__ void __launch_bounds__(256)
sample_kernel(const float* __restrict__ scorem, const int* __restrict__ next_rel,
              float* __restrict__ logp_out, float* __restrict__ loss_out,
              int* __restrict__ action_out, int* __restrict__ chosen_out)
{
    __shared__ float red[256];
    __shared__ int   redi[256];
    __shared__ int   act_s;
    __shared__ float lse_s;

    int b = blockIdx.x, a = threadIdx.x;
    float s = scorem[(size_t)b * 256 + a];

    // JAX: key(42) -> (0,42); counts = iota(B*A) split into halves
    unsigned idx = (unsigned)(b * 256 + a);
    const unsigned half = 131072u;  // (1024*256)/2
    unsigned x0, x1; int take;
    if (idx < half) { x0 = idx;        x1 = idx + half; take = 0; }
    else            { x0 = idx - half; x1 = idx;        take = 1; }
    threefry2x32(0u, 42u, x0, x1);
    unsigned bits = take ? x1 : x0;
    float f = __uint_as_float(0x3f800000u | (bits >> 9)) - 1.0f;
    const float tiny = 1.1754943508222875e-38f;
    float u = f * (1.0f - tiny) + tiny;
    float g = -logf(-logf(u));
    float pert = s + g;

    // argmax(s + gumbel), first-index tie break
    red[a] = pert; redi[a] = a; __syncthreads();
    for (int st = 128; st > 0; st >>= 1) {
        if (a < st) {
            float o = red[a + st]; int oi = redi[a + st];
            if (o > red[a] || (o == red[a] && oi < redi[a])) { red[a] = o; redi[a] = oi; }
        }
        __syncthreads();
    }
    if (a == 0) act_s = redi[0];
    __syncthreads();

    // log-softmax of masked scores
    red[a] = s; __syncthreads();
    for (int st = 128; st > 0; st >>= 1) {
        if (a < st) red[a] = fmaxf(red[a], red[a + st]);
        __syncthreads();
    }
    float mx = red[0]; __syncthreads();
    red[a] = expf(s - mx); __syncthreads();
    for (int st = 128; st > 0; st >>= 1) {
        if (a < st) red[a] += red[a + st];
        __syncthreads();
    }
    if (a == 0) lse_s = mx + logf(red[0]);
    __syncthreads();

    float lp = s - lse_s;
    logp_out[(size_t)b * 256 + a] = lp;
    if (a == act_s) loss_out[b] = -lp;
    if (a == 0) {
        action_out[b] = act_s;
        chosen_out[b] = next_rel[(size_t)b * 256 + act_s];
    }
}

// ---------------------------------------------------------------------------
// Host orchestration
// ---------------------------------------------------------------------------
extern "C" void kernel_launch(void* const* d_in, const int* in_sizes, int n_in,
                              void* d_out, int out_size, void* d_ws, size_t ws_size,
                              hipStream_t stream)
{
    (void)in_sizes; (void)n_in; (void)out_size; (void)ws_size;

    const int*   next_rel  = (const int*)d_in[0];
    const int*   next_ent  = (const int*)d_in[1];
    const int*   prev_rel  = (const int*)d_in[2];
    const int*   cur_ent   = (const int*)d_in[3];
    const float* query     = (const float*)d_in[5];
    const float* prev_h0   = (const float*)d_in[6];
    const float* prev_c0   = (const float*)d_in[7];
    const float* prev_h1   = (const float*)d_in[8];
    const float* prev_c1   = (const float*)d_in[9];
    const float* rel_table = (const float*)d_in[10];
    const float* ent_table = (const float*)d_in[11];
    const float* Wx0       = (const float*)d_in[12];
    const float* Wh0       = (const float*)d_in[13];
    const float* b0        = (const float*)d_in[14];
    const float* Wx1       = (const float*)d_in[15];
    const float* Wh1       = (const float*)d_in[16];
    const float* b1        = (const float*)d_in[17];
    const float* W1d       = (const float*)d_in[18];
    const float* b1d       = (const float*)d_in[19];
    const float* W2d       = (const float*)d_in[20];
    const float* b2d       = (const float*)d_in[21];

    // workspace layout (20 MiB, buffers reused via stream ordering)
    char* ws = (char*)d_ws;
    unsigned short* Wt  = (unsigned short*)(ws);                        // 4 MiB max [N,K] bf16
    float*          Z   = (float*)         (ws + (size_t)4  * 1048576); // 8 MiB  [1024,2048] f32
    unsigned short* X   = (unsigned short*)(ws + (size_t)12 * 1048576); // 2 MiB  [1024,1024] bf16
    unsigned short* SQ  = (unsigned short*)(ws + (size_t)14 * 1048576); // 2 MiB  [1024,1024] bf16
    unsigned short* HID = (unsigned short*)(ws + (size_t)16 * 1048576); // 1 MiB  [1024,512]  bf16
    float*          MLP = (float*)         (ws + (size_t)17 * 1048576); // 2 MiB  [1024,512]  f32
    float*          SCM = (float*)         (ws + (size_t)19 * 1048576); // 1 MiB  [1024,256]  f32

    // output layout (flat concat in return order)
    float* out      = (float*)d_out;
    float* o_loss   = out;                 // [1024]
    float* o_h0     = out + 1024;          // [1024,512]
    float* o_c0     = out + 525312;
    float* o_h1     = out + 1049600;
    float* o_c1     = out + 1573888;
    float* o_logp   = out + 2098176;       // [1024,256]
    int*   o_action = (int*)out + 2360320; // [1024] int32
    int*   o_chosen = (int*)out + 2361344; // [1024] int32
    float* o_prelim = out + 2362368;       // [1024,256]

    dim3 tb(32, 8);

    // build bf16 activations + gathered embeddings
    gather_kernel<<<1024, 256, 0, stream>>>(prev_rel, cur_ent, rel_table, ent_table,
                                            prev_h0, query, X, SQ);

    // ---- LSTM layer 0: z0 = [prev_act|h0_prev] @ [Wx0;Wh0] + b0
    transpose_to_bf16<<<dim3(32, 64), tb, 0, stream>>>(Wx0, Wh0, 512, 1024, 2048, Wt);
    wmma_gemm_bf16<<<dim3(32, 8), 128, 0, stream>>>(X, Wt, b0, Z, nullptr,
                                                    1024, 2048, 1024, 0);
    lstm_act_kernel<<<1024, 256, 0, stream>>>(Z, prev_c0, prev_h1, o_h0, o_c0, X);

    // ---- LSTM layer 1
    transpose_to_bf16<<<dim3(32, 64), tb, 0, stream>>>(Wx1, Wh1, 512, 1024, 2048, Wt);
    wmma_gemm_bf16<<<dim3(32, 8), 128, 0, stream>>>(X, Wt, b1, Z, nullptr,
                                                    1024, 2048, 1024, 0);
    lstm_act_kernel<<<1024, 256, 0, stream>>>(Z, prev_c1, nullptr, o_h1, o_c1, SQ);

    // ---- MLP layer 1: hidden = relu(SQ @ W1d + b1d)
    transpose_to_bf16<<<dim3(32, 16), tb, 0, stream>>>(W1d, nullptr, 1024, 1024, 512, Wt);
    wmma_gemm_bf16<<<dim3(8, 8), 128, 0, stream>>>(SQ, Wt, b1d, nullptr, HID,
                                                   1024, 512, 1024, 1);

    // ---- MLP layer 2: mlp_out = relu(hidden @ W2d + b2d)
    transpose_to_bf16<<<dim3(16, 16), tb, 0, stream>>>(W2d, nullptr, 512, 512, 512, Wt);
    wmma_gemm_bf16<<<dim3(8, 8), 128, 0, stream>>>(HID, Wt, b2d, MLP, nullptr,
                                                   1024, 512, 512, 1);

    // ---- candidate scores (L2-bound gather-dot)
    scores_kernel<<<1024, 256, 0, stream>>>(next_rel, next_ent, rel_table, ent_table,
                                            MLP, o_prelim, SCM);

    // ---- categorical sample + log-softmax + loss
    sample_kernel<<<1024, 256, 0, stream>>>(SCM, next_rel, o_logp, o_loss,
                                            o_action, o_chosen);
}
